// AttentionFlowLayer_19284403159693
// MI455X (gfx1250) — matrix operand order repaired and checked by reading, not compiled, not run on hardware
//
#include <hip/hip_runtime.h>
#include <hip/hip_bf16.h>

// ---------------------------------------------------------------------------
// BiDAF attention-flow layer for MI455X (gfx1250, wave32, WMMA).
// B=32, Lc=512, Lq=64, D=1024.  Memory-bound (~330MB min traffic @23.3TB/s
// => ~14us); GEMMs run on v_wmma_f32_16x16x32_bf16 (f32 accumulate) so the
// matrix math stays far below the bandwidth roofline.
// ---------------------------------------------------------------------------

typedef __attribute__((ext_vector_type(16))) __bf16 v16bf;
typedef __attribute__((ext_vector_type(8)))  __bf16 v8bf;
typedef __attribute__((ext_vector_type(8)))  float  v8f;

constexpr int NB    = 32;    // batch
constexpr int LC    = 512;   // context length
constexpr int LQ    = 64;    // query length
constexpr int DIM   = 1024;  // feature dim
constexpr int KC    = 128;   // K-chunk staged in LDS
constexpr int CTILE = 64;    // c rows per workgroup

// ---- WMMA fragment builders (contiguous-k order, same in A and B, wave32) --------
// A (16x32 bf16): lane half selects K-subblocks; per lane two contiguous
// 8-element runs: K = kbase + half*16 + [0..7] and K = kbase + half*16 + 8 + [0..7]
__device__ __forceinline__ v16bf load_a_frag(const __bf16* base, int stride,
                                             int row0, int kbase, int half, int lm) {
  const __bf16* p = base + (row0 + lm) * stride + kbase + half * 16;
  v8bf lo = *(const v8bf*)p;
  v8bf hi = *(const v8bf*)(p + 8);
  v16bf a;
#pragma unroll
  for (int i = 0; i < 8; ++i) { a[i] = lo[i]; a[i + 8] = hi[i]; }
  return a;
}

// B (32x16 bf16), with B[k][n] stored contiguous-in-k at base[(col0+n)*stride + k]:
// per lane one contiguous 16-element run starting at kbase + half*16.
__device__ __forceinline__ v16bf load_b_frag(const __bf16* base, int stride,
                                             int col0, int kbase, int half, int lm) {
  return *(const v16bf*)(base + (col0 + lm) * stride + kbase + half * 16);
}

__device__ __forceinline__ v8f wmma_bf16(v16bf a, v16bf b, v8f c) {
  return __builtin_amdgcn_wmma_f32_16x16x32_bf16(false, a, false, b,
                                                 (short)0, c, false, false);
}

// ---------------------------------------------------------------------------
// Kernel 1: s0 = c @ proj_c  [NB*LC],  s1 = q @ proj_q  [NB*LQ]
// One wave per row.
// ---------------------------------------------------------------------------
__global__ __launch_bounds__(256)
void gemv_kernel(const float* __restrict__ c, const float* __restrict__ q,
                 const float* __restrict__ pc, const float* __restrict__ pq,
                 float* __restrict__ s0, float* __restrict__ s1) {
  int row  = blockIdx.x * 8 + (threadIdx.x >> 5);
  int lane = threadIdx.x & 31;
  const float* vec;
  const float* w;
  float* dst;
  if (row < NB * LC) {
    vec = c + (size_t)row * DIM; w = pc; dst = s0 + row;
  } else {
    int r2 = row - NB * LC;
    vec = q + (size_t)r2 * DIM; w = pq; dst = s1 + r2;
  }
  float s = 0.f;
#pragma unroll 4
  for (int d = lane; d < DIM; d += 32) s += vec[d] * w[d];
#pragma unroll
  for (int o = 16; o >= 1; o >>= 1) s += __shfl_xor(s, o, 32);
  if (lane == 0) *dst = s;
}

// ---------------------------------------------------------------------------
// Kernel 2: per (batch, 64-row c-tile): sim via WMMA, softmax, c2q via WMMA,
// write out sections 0 (c), 1 (c2q), 3 (c*c2q); store row-max(sim) for q2c.
// ---------------------------------------------------------------------------
__global__ __launch_bounds__(256)
void attn_kernel(const float* __restrict__ c, const float* __restrict__ q,
                 const float* __restrict__ proj_cq,
                 const float* __restrict__ s0g, const float* __restrict__ s1g,
                 float* __restrict__ rowmax_g, float* __restrict__ out) {
  __shared__ __bf16 qbuf[LQ * KC];      // ph1: [qr][k]; ph3: [d][qr]
  __shared__ __bf16 cbuf[CTILE * KC];   // bf16(c * proj_cq)
  __shared__ float  simbuf[CTILE * LQ];
  __shared__ __bf16 psbuf[CTILE * LQ];  // softmax probs, bf16
  __shared__ float  pcq_s[DIM];
  __shared__ float  s0s[CTILE];
  __shared__ float  s1s[LQ];

  const int tid  = threadIdx.x;
  const int w    = tid >> 5;
  const int lane = tid & 31;
  const int half = lane >> 4;
  const int lm   = lane & 15;
  const int bat  = blockIdx.x >> 3;
  const int ct0  = (blockIdx.x & 7) * CTILE;

  for (int i = tid; i < DIM; i += 256) pcq_s[i] = proj_cq[i];
  if (tid < CTILE)               s0s[tid]         = s0g[bat * LC + ct0 + tid];
  else if (tid < CTILE + LQ)     s1s[tid - CTILE] = s1g[bat * LQ + (tid - CTILE)];
  __syncthreads();

  // --- Phase 1: sim = (c*pcq) @ q^T, tiled 16x16 per wave (2 tiles/wave) ---
  const int mi  = (2 * w) >> 2;        // both tiles of a wave share mi
  const int ni0 = (2 * w) & 3;
  const int ni1 = (2 * w + 1) & 3;
  v8f acc0 = {}, acc1 = {};

  for (int kt = 0; kt < DIM / KC; ++kt) {
    if (kt) __syncthreads();           // previous chunk's WMMA reads done
#pragma unroll
    for (int i = 0; i < (LQ * KC) / 256; ++i) {
      int e = i * 256 + tid;
      int qr = e >> 7, d = e & (KC - 1);
      qbuf[e] = (__bf16)q[((size_t)bat * LQ + qr) * DIM + kt * KC + d];
    }
#pragma unroll
    for (int i = 0; i < (CTILE * KC) / 256; ++i) {
      int e = i * 256 + tid;
      int cr = e >> 7, d = e & (KC - 1);
      int kd = kt * KC + d;
      cbuf[e] = (__bf16)(c[((size_t)bat * LC + ct0 + cr) * DIM + kd] * pcq_s[kd]);
    }
    if (kt + 1 < DIM / KC)             // hint next chunk of c into cache
      __builtin_prefetch(&c[((size_t)bat * LC + ct0 + (tid >> 2)) * DIM +
                            (kt + 1) * KC + (tid & 3) * 32], 0, 1);
    __syncthreads();
#pragma unroll
    for (int ks = 0; ks < KC / 32; ++ks) {
      v16bf a  = load_a_frag(cbuf, KC, mi * 16, ks * 32, half, lm);
      v16bf b0 = load_b_frag(qbuf, KC, ni0 * 16, ks * 32, half, lm);
      v16bf b1 = load_b_frag(qbuf, KC, ni1 * 16, ks * 32, half, lm);
      acc0 = wmma_bf16(a, b0, acc0);
      acc1 = wmma_bf16(a, b1, acc1);
    }
  }
  // epilogue: sim = s2 + s0[row] + s1[col]   (C/D layout: vgpr r -> M=r+8*half)
  {
    float s1c0 = s1s[ni0 * 16 + lm];
    float s1c1 = s1s[ni1 * 16 + lm];
#pragma unroll
    for (int r = 0; r < 8; ++r) {
      int m = mi * 16 + r + 8 * half;
      float s0v = s0s[m];
      simbuf[m * LQ + ni0 * 16 + lm] = acc0[r] + s0v + s1c0;
      simbuf[m * LQ + ni1 * 16 + lm] = acc1[r] + s0v + s1c1;
    }
  }
  __syncthreads();

  // --- Phase 2: softmax over Lq (4 lanes per row), store rowmax ------------
  {
    int row = tid >> 2, quad = tid & 3;
    const float* srow = &simbuf[row * LQ + quad * 16];
    float mx = -3.0e38f;
#pragma unroll
    for (int i = 0; i < 16; ++i) mx = fmaxf(mx, srow[i]);
    mx = fmaxf(mx, __shfl_xor(mx, 1, 32));
    mx = fmaxf(mx, __shfl_xor(mx, 2, 32));
    if (quad == 0) rowmax_g[bat * LC + ct0 + row] = mx;
    float e[16], s = 0.f;
#pragma unroll
    for (int i = 0; i < 16; ++i) { e[i] = __expf(srow[i] - mx); s += e[i]; }
    s += __shfl_xor(s, 1, 32);
    s += __shfl_xor(s, 2, 32);
    float inv = 1.f / s;
    __bf16* prow = &psbuf[row * LQ + quad * 16];
#pragma unroll
    for (int i = 0; i < 16; ++i) prow[i] = (__bf16)(e[i] * inv);
  }
  __syncthreads();

  // --- Phase 3: c2q = P @ q ; write out sections 0, 1, 3 -------------------
  for (int nt = 0; nt < DIM / KC; ++nt) {
    __syncthreads();                   // previous chunk's WMMA reads done
    // stage q chunk transposed: qbuf viewed as qt[KC][LQ]
#pragma unroll
    for (int i = 0; i < (LQ * KC) / 256; ++i) {
      int e = i * 256 + tid;
      int qr = e >> 7, d = e & (KC - 1);
      qbuf[d * LQ + qr] = (__bf16)q[((size_t)bat * LQ + qr) * DIM + nt * KC + d];
    }
    __syncthreads();
#pragma unroll
    for (int tt = 0; tt < 4; ++tt) {
      int tile = w * 4 + tt;           // 32 tiles: 4 (M) x 8 (N16)
      int tmi = tile >> 3;
      int tni = tile & 7;
      v8f acc = {};
#pragma unroll
      for (int ks = 0; ks < LQ / 32; ++ks) {
        v16bf a  = load_a_frag(psbuf, LQ, tmi * 16, ks * 32, half, lm);
        v16bf bm = load_b_frag(qbuf,  LQ, tni * 16, ks * 32, half, lm);
        acc = wmma_bf16(a, bm, acc);
      }
      int d = nt * KC + tni * 16 + lm;
#pragma unroll
      for (int r = 0; r < 8; ++r) {
        int row = ct0 + tmi * 16 + r + 8 * half;
        size_t cidx = ((size_t)bat * LC + row) * DIM + d;
        float cv  = c[cidx];
        float c2q = acc[r];
        size_t ob = ((size_t)bat * LC + row) * (4 * DIM);
        out[ob + d]           = cv;        // section 0: c
        out[ob + DIM + d]     = c2q;       // section 1: c2q
        out[ob + 3 * DIM + d] = cv * c2q;  // section 3: c * c2q
      }
    }
  }
}

// ---------------------------------------------------------------------------
// Kernel 3: q2c[b,:] = softmax_Lc(rowmax) @ c[b]   (one block per batch)
// ---------------------------------------------------------------------------
__global__ __launch_bounds__(256)
void q2c_kernel(const float* __restrict__ c, const float* __restrict__ rowmax_g,
                float* __restrict__ q2c_g) {
  __shared__ float wgt[LC];
  __shared__ float red[8];
  int bat = blockIdx.x, tid = threadIdx.x;
  float l0 = rowmax_g[bat * LC + tid];
  float l1 = rowmax_g[bat * LC + 256 + tid];
  float mx = fmaxf(l0, l1);
#pragma unroll
  for (int o = 16; o >= 1; o >>= 1) mx = fmaxf(mx, __shfl_xor(mx, o, 32));
  if ((tid & 31) == 0) red[tid >> 5] = mx;
  __syncthreads();
  float gmx = red[0];
#pragma unroll
  for (int i = 1; i < 8; ++i) gmx = fmaxf(gmx, red[i]);
  float e0 = __expf(l0 - gmx), e1 = __expf(l1 - gmx);
  float s = e0 + e1;
#pragma unroll
  for (int o = 16; o >= 1; o >>= 1) s += __shfl_xor(s, o, 32);
  __syncthreads();
  if ((tid & 31) == 0) red[tid >> 5] = s;
  __syncthreads();
  float tot = 0.f;
#pragma unroll
  for (int i = 0; i < 8; ++i) tot += red[i];
  float inv = 1.f / tot;
  wgt[tid]       = e0 * inv;
  wgt[tid + 256] = e1 * inv;
  __syncthreads();
  float acc[4] = {0.f, 0.f, 0.f, 0.f};
  for (int cc = 0; cc < LC; ++cc) {
    float wv = wgt[cc];
    const float* crow = c + ((size_t)bat * LC + cc) * DIM;
#pragma unroll
    for (int j = 0; j < 4; ++j) acc[j] += wv * crow[tid + j * 256];
  }
#pragma unroll
  for (int j = 0; j < 4; ++j) q2c_g[bat * DIM + tid + j * 256] = acc[j];
}

// ---------------------------------------------------------------------------
// Kernel 4: out section 2 = c * q2c (broadcast over Lc). Pure streaming.
// ---------------------------------------------------------------------------
__global__ __launch_bounds__(256)
void finalize_kernel(const float* __restrict__ c, const float* __restrict__ q2c_g,
                     float* __restrict__ out) {
  size_t i = ((size_t)blockIdx.x * 256 + threadIdx.x) * 4;
  float4 cv = *(const float4*)(c + i);
  size_t d   = i & (DIM - 1);
  size_t bl  = i >> 10;   // b*LC + row
  size_t bat = bl >> 9;
  float4 qv = *(const float4*)(q2c_g + bat * DIM + d);
  float4 r{cv.x * qv.x, cv.y * qv.y, cv.z * qv.z, cv.w * qv.w};
  *(float4*)(out + bl * (4 * DIM) + 2 * DIM + d) = r;
}

// ---------------------------------------------------------------------------
extern "C" void kernel_launch(void* const* d_in, const int* in_sizes, int n_in,
                              void* d_out, int out_size, void* d_ws, size_t ws_size,
                              hipStream_t stream) {
  const float* c       = (const float*)d_in[0];
  const float* q       = (const float*)d_in[1];
  const float* proj_c  = (const float*)d_in[2];
  const float* proj_q  = (const float*)d_in[3];
  const float* proj_cq = (const float*)d_in[4];
  float* out = (float*)d_out;

  float* ws     = (float*)d_ws;
  float* s0     = ws;                        // NB*LC      = 16384
  float* s1     = ws + NB * LC;              // NB*LQ      =  2048
  float* rowmax = s1 + NB * LQ;              // NB*LC      = 16384
  float* q2c    = rowmax + NB * LC;          // NB*DIM     = 32768

  gemv_kernel<<<(NB * LC + NB * LQ) / 8, 256, 0, stream>>>(c, q, proj_c, proj_q, s0, s1);
  attn_kernel<<<NB * (LC / CTILE), 256, 0, stream>>>(c, q, proj_cq, s0, s1, rowmax, out);
  q2c_kernel<<<NB, 256, 0, stream>>>(c, rowmax, q2c);
  finalize_kernel<<<(NB * LC * DIM) / (256 * 4), 256, 0, stream>>>(c, q2c, out);
}
